// GraphConvolution_52647709114619
// MI455X (gfx1250) — matrix operand order repaired and checked by reading, not compiled, not run on hardware
//
#include <hip/hip_runtime.h>

typedef __attribute__((ext_vector_type(16))) _Float16 v16h;
typedef __attribute__((ext_vector_type(8)))  _Float16 v8h;
typedef __attribute__((ext_vector_type(8)))  float    v8f;
typedef __attribute__((ext_vector_type(4)))  float    f32x4;

#define R_ 5
#define M_ 4096
#define N_ 4096
#define D_ 256
#define H_ 64

__device__ __forceinline__ float safe_inv_sqrt(float x) {
    return x > 0.0f ? rsqrtf(x) : 0.0f;
}

// ---------------------------------------------------------------------------
// Zero a float region (r_sums | z_u_full | z_v_full are contiguous in ws).
// ---------------------------------------------------------------------------
__global__ void zero_kernel(float* __restrict__ p, int n) {
    int tid = blockIdx.x * blockDim.x + threadIdx.x;
    if (tid < n) p[tid] = 0.0f;
}

// ---------------------------------------------------------------------------
// One streaming pass over support: row sums (over n) -> c_sums[R,M],
// column sums (over m) -> r_sums[R,N] (pre-zeroed, global f32 atomics).
// Block: 256 threads handles (r, 128 rows, all 4096 cols).
// ---------------------------------------------------------------------------
__global__ void sums_kernel(const float* __restrict__ supp,
                            float* __restrict__ c_sums,
                            float* __restrict__ r_sums) {
    __shared__ float lds_row[128];
    const int r  = blockIdx.y;
    const int m0 = blockIdx.x * 128;
    const int t  = threadIdx.x;              // 0..255
    if (t < 128) lds_row[t] = 0.0f;
    __syncthreads();

    const int c0 = t * 16;                   // this thread's 16 columns
    float col_acc[16];
#pragma unroll
    for (int i = 0; i < 16; ++i) col_acc[i] = 0.0f;

    const float* base = supp + ((size_t)r * M_ + m0) * N_ + c0;
    for (int mi = 0; mi < 128; ++mi) {
        const f32x4* p = (const f32x4*)(base + (size_t)mi * N_);
        float s = 0.0f;
#pragma unroll
        for (int j = 0; j < 4; ++j) {
            f32x4 q = p[j];
#pragma unroll
            for (int k = 0; k < 4; ++k) {
                float x = q[k];
                col_acc[j * 4 + k] += x;
                s += x;
            }
        }
        // wave32 reduction of the row partial, then one LDS atomic per wave
#pragma unroll
        for (int off = 16; off > 0; off >>= 1) s += __shfl_down(s, off, 32);
        if ((t & 31) == 0) atomicAdd(&lds_row[mi], s);
    }
    __syncthreads();
    if (t < 128) c_sums[(size_t)r * M_ + m0 + t] = lds_row[t];
#pragma unroll
    for (int i = 0; i < 16; ++i)
        atomicAdd(r_sums + (size_t)r * N_ + c0 + i, col_acc[i]);
}

// ---------------------------------------------------------------------------
// cum_w = cumsum(u_weight, axis=0), stored transposed as f16: cwT[r,h,d].
// ---------------------------------------------------------------------------
__global__ void cumw_kernel(const float* __restrict__ w, _Float16* __restrict__ cwT) {
    int tid = blockIdx.x * blockDim.x + threadIdx.x;   // D*H threads
    if (tid >= D_ * H_) return;
    int d = tid / H_, h = tid % H_;
    float acc = 0.0f;
#pragma unroll
    for (int r = 0; r < R_; ++r) {
        acc += w[((size_t)r * D_ + d) * H_ + h];
        cwT[((size_t)r * H_ + h) * D_ + d] = (_Float16)acc;
    }
}

// ---------------------------------------------------------------------------
// tmpT[r,h,row] = (f16)( safe_inv_sqrt(sums[r,row]) * sum_d feat[row,d]*cwT[r,h,d] )
// WMMA f16 GEMM: wave = 16 rows x 64 h, K loop over D in steps of 32.
// ---------------------------------------------------------------------------
__global__ void feat_gemm_kernel(const float* __restrict__ feat,     // [4096,D]
                                 const _Float16* __restrict__ cwT,   // [R,H,D]
                                 const float* __restrict__ sums,     // [R,4096]
                                 _Float16* __restrict__ outT) {      // [R,H,4096]
    const int r     = blockIdx.y;
    const int lane  = threadIdx.x & 31;
    const int w     = threadIdx.x >> 5;
    const int rbase = blockIdx.x * 128 + w * 16;
    const int mrow  = rbase + (lane & 15);
    const int k0    = (lane >> 4) * 16;      // A order used here (B matches): 16 contiguous K halves k0..k0+15

    v8f zero = {};
    v8f acc[4];
#pragma unroll
    for (int i = 0; i < 4; ++i) acc[i] = zero;

    for (int d0 = 0; d0 < D_; d0 += 32) {
        const float* arow = feat + (size_t)mrow * D_ + d0;
        const f32x4* lo = (const f32x4*)(arow + k0);
        const f32x4* hi = (const f32x4*)(arow + k0 + 8);
        v16h a;
#pragma unroll
        for (int j = 0; j < 2; ++j) {
            f32x4 ql = lo[j], qh = hi[j];
#pragma unroll
            for (int k = 0; k < 4; ++k) {
                a[j * 4 + k]     = (_Float16)ql[k];
                a[8 + j * 4 + k] = (_Float16)qh[k];
            }
        }
        const int kb = (lane >> 4) * 16;     // B layout: 16 contiguous K halves
#pragma unroll
        for (int ht = 0; ht < 4; ++ht) {
            const v16h b = *(const v16h*)(cwT + ((size_t)r * H_ + ht * 16 + (lane & 15)) * D_ + d0 + kb);
            acc[ht] = __builtin_amdgcn_wmma_f32_16x16x32_f16(false, a, false, b, (short)0, acc[ht], false, false);
        }
    }
    // epilogue: scale per row by safe 1/sqrt(degree), store transposed f16
#pragma unroll
    for (int i = 0; i < 8; ++i) {
        const int row   = rbase + i + (lane >> 4) * 8;   // C layout rows
        const float inv = safe_inv_sqrt(sums[(size_t)r * 4096 + row]);
#pragma unroll
        for (int ht = 0; ht < 4; ++ht) {
            const int h = ht * 16 + (lane & 15);
            outT[((size_t)r * H_ + h) * 4096 + row] = (_Float16)(acc[ht][i] * inv);
        }
    }
}

// ---------------------------------------------------------------------------
// z_u_full[m,h] += sum_n (c_inv[r,m]*supp[r,m,n]) * tvT[r,h,n]
// A converted f32->f16 in registers (row-major, single use); B from L2.
// Grid: (M/128, R, 4 n-chunks); accumulate via global f32 atomics.
// ---------------------------------------------------------------------------
__global__ void zu_gemm_kernel(const float* __restrict__ supp,    // [R,M,N]
                               const _Float16* __restrict__ tvT,  // [R,H,N] (r_inv folded)
                               const float* __restrict__ c_sums,  // [R,M]
                               float* __restrict__ z_u) {         // [M,H] pre-zeroed
    const int lane  = threadIdx.x & 31;
    const int w     = threadIdx.x >> 5;
    const int mbase = blockIdx.x * 128 + w * 16;
    const int mrow  = mbase + (lane & 15);
    const int k0    = (lane >> 4) * 16;
    const int kb    = (lane >> 4) * 16;
    const int r     = blockIdx.y;

    v8f zero = {};
    v8f acc[4];
#pragma unroll
    for (int i = 0; i < 4; ++i) acc[i] = zero;

    const float  cinv = safe_inv_sqrt(c_sums[(size_t)r * M_ + mrow]);
    const float* arow = supp + ((size_t)r * M_ + mrow) * N_;

    const int nbeg = blockIdx.z * (N_ / 4), nend = nbeg + (N_ / 4);
    for (int n0 = nbeg; n0 < nend; n0 += 32) {
        __builtin_prefetch(arow + n0 + 256, 0, 0);   // -> global_prefetch_b8
        const f32x4* lo = (const f32x4*)(arow + n0 + k0);
        const f32x4* hi = (const f32x4*)(arow + n0 + k0 + 8);
        v16h a;
#pragma unroll
        for (int j = 0; j < 2; ++j) {
            f32x4 ql = lo[j], qh = hi[j];
#pragma unroll
            for (int k = 0; k < 4; ++k) {
                a[j * 4 + k]     = (_Float16)(ql[k] * cinv);
                a[8 + j * 4 + k] = (_Float16)(qh[k] * cinv);
            }
        }
#pragma unroll
        for (int ht = 0; ht < 4; ++ht) {
            const v16h b = *(const v16h*)(tvT + ((size_t)r * H_ + ht * 16 + (lane & 15)) * N_ + n0 + kb);
            acc[ht] = __builtin_amdgcn_wmma_f32_16x16x32_f16(false, a, false, b, (short)0, acc[ht], false, false);
        }
    }
#pragma unroll
    for (int i = 0; i < 8; ++i) {
        const int row = mbase + i + (lane >> 4) * 8;
#pragma unroll
        for (int ht = 0; ht < 4; ++ht)
            atomicAdd(&z_u[(size_t)row * H_ + ht * 16 + (lane & 15)], acc[ht][i]);
    }
}

// ---------------------------------------------------------------------------
// z_v_full[n,h] += sum_m (r_inv[r,n]*supp[r,m,n]) * tuT[r,h,m]
// Needs supp^T: stage scaled-f16 tiles through LDS with transpose-on-store.
// Grid: (N/128, R, 4 m-chunks); accumulate via global f32 atomics.
// ---------------------------------------------------------------------------
__global__ void zv_gemm_kernel(const float* __restrict__ supp,    // [R,M,N]
                               const _Float16* __restrict__ tuT,  // [R,H,M] (c_inv folded)
                               const float* __restrict__ r_sums,  // [R,N]
                               float* __restrict__ z_v) {         // [N,H] pre-zeroed
    __shared__ _Float16 ldsA[128 * 32];   // [n_local][m_local] f16, 8 KB
    __shared__ float    rinv_s[128];
    const int lane  = threadIdx.x & 31;
    const int w     = threadIdx.x >> 5;
    const int r     = blockIdx.y;
    const int n0blk = blockIdx.x * 128;

    if (threadIdx.x < 128)
        rinv_s[threadIdx.x] = safe_inv_sqrt(r_sums[(size_t)r * N_ + n0blk + threadIdx.x]);
    __syncthreads();

    v8f zero = {};
    v8f acc[4];
#pragma unroll
    for (int i = 0; i < 4; ++i) acc[i] = zero;

    const int nloc = w * 16 + (lane & 15);
    const int k0   = (lane >> 4) * 16;
    const int kb   = (lane >> 4) * 16;
    const int t    = threadIdx.x;
    const int smi  = t >> 3;              // staging m row 0..31
    const int scb  = (t & 7) * 16;        // staging n-col base

    const int mbeg = blockIdx.z * (M_ / 4), mend = mbeg + (M_ / 4);
    for (int m0 = mbeg; m0 < mend; m0 += 32) {
        // stage 32(m) x 128(n) f32 tile -> scaled f16, transposed in LDS
        const f32x4* sp = (const f32x4*)(supp + ((size_t)r * M_ + m0 + smi) * N_ + n0blk + scb);
#pragma unroll
        for (int j4 = 0; j4 < 4; ++j4) {
            f32x4 q = sp[j4];
#pragma unroll
            for (int k = 0; k < 4; ++k) {
                const int nl = scb + j4 * 4 + k;
                ldsA[nl * 32 + smi] = (_Float16)(q[k] * rinv_s[nl]);
            }
        }
        __syncthreads();
        // A' fragment (rows = n, K = m) from LDS, 16-bit A layout
        v8h a_lo = *(const v8h*)(ldsA + nloc * 32 + k0);
        v8h a_hi = *(const v8h*)(ldsA + nloc * 32 + k0 + 8);
        v16h a;
#pragma unroll
        for (int i = 0; i < 8; ++i) { a[i] = a_lo[i]; a[8 + i] = a_hi[i]; }
#pragma unroll
        for (int ht = 0; ht < 4; ++ht) {
            const v16h b = *(const v16h*)(tuT + ((size_t)r * H_ + ht * 16 + (lane & 15)) * M_ + m0 + kb);
            acc[ht] = __builtin_amdgcn_wmma_f32_16x16x32_f16(false, a, false, b, (short)0, acc[ht], false, false);
        }
        __syncthreads();
    }
#pragma unroll
    for (int i = 0; i < 8; ++i) {
        const int row = n0blk + w * 16 + i + (lane >> 4) * 8;
#pragma unroll
        for (int ht = 0; ht < 4; ++ht)
            atomicAdd(&z_v[(size_t)row * H_ + ht * 16 + (lane & 15)], acc[ht][i]);
    }
}

// ---------------------------------------------------------------------------
// Gather + bias + ReLU: out = [relu(z_u_full[u[b]]+bias) ; relu(z_v_full[v[b]]+bias)]
// ---------------------------------------------------------------------------
__global__ void gather_kernel(const float* __restrict__ z_u_full,
                              const float* __restrict__ z_v_full,
                              const int* __restrict__ u,
                              const int* __restrict__ v,
                              const float* __restrict__ bias,
                              float* __restrict__ out) {
    int tid = blockIdx.x * blockDim.x + threadIdx.x;
    const int half = 4096 * H_;
    if (tid >= 2 * half) return;
    const bool is_v = tid >= half;
    const int lid = is_v ? tid - half : tid;
    const int b = lid >> 6, h = lid & (H_ - 1);
    const int idx = is_v ? v[b] : u[b];
    const float* full = is_v ? z_v_full : z_u_full;
    const float val = full[(size_t)idx * H_ + h] + bias[h];
    out[tid] = val > 0.0f ? val : 0.0f;
}

// ---------------------------------------------------------------------------
// Workspace layout (bytes):
//   0        r_sums   [R,N] f32      81920   } contiguous zeroed prefix
//   81920    z_u_full [M,H] f32    1048576   }
//   1130496  z_v_full [N,H] f32    1048576   }
//   2179072  c_sums   [R,M] f32      81920
//   2260992  cwT      [R,H,D] f16   163840
//   2424832  tvT      [R,H,N] f16  2621440
//   5046272  tuT      [R,H,M] f16  2621440   (total ~7.7 MB)
// ---------------------------------------------------------------------------
extern "C" void kernel_launch(void* const* d_in, const int* in_sizes, int n_in,
                              void* d_out, int out_size, void* d_ws, size_t ws_size,
                              hipStream_t stream) {
    (void)in_sizes; (void)n_in; (void)out_size; (void)ws_size;
    const float* u_feat = (const float*)d_in[0];
    const float* v_feat = (const float*)d_in[1];
    const int*   u_idx  = (const int*)d_in[2];
    const int*   v_idx  = (const int*)d_in[3];
    const float* supp   = (const float*)d_in[4];
    const float* u_w    = (const float*)d_in[5];
    const float* u_bias = (const float*)d_in[6];
    float* out = (float*)d_out;

    char* ws = (char*)d_ws;
    float*    r_sums = (float*)(ws + 0);
    float*    z_u    = (float*)(ws + 81920);
    float*    z_v    = (float*)(ws + 1130496);
    float*    c_sums = (float*)(ws + 2179072);
    _Float16* cwT    = (_Float16*)(ws + 2260992);
    _Float16* tvT    = (_Float16*)(ws + 2424832);
    _Float16* tuT    = (_Float16*)(ws + 5046272);

    const int nz = (81920 + 1048576 + 1048576) / 4;   // zero r_sums|z_u|z_v
    zero_kernel<<<(nz + 255) / 256, 256, 0, stream>>>(r_sums, nz);

    sums_kernel<<<dim3(M_ / 128, R_), 256, 0, stream>>>(supp, c_sums, r_sums);
    cumw_kernel<<<(D_ * H_ + 255) / 256, 256, 0, stream>>>(u_w, cwT);
    feat_gemm_kernel<<<dim3(N_ / 128, R_), 256, 0, stream>>>(v_feat, cwT, r_sums, tvT);
    feat_gemm_kernel<<<dim3(M_ / 128, R_), 256, 0, stream>>>(u_feat, cwT, c_sums, tuT);
    zu_gemm_kernel<<<dim3(M_ / 128, R_, 4), 256, 0, stream>>>(supp, tvT, c_sums, z_u);
    zv_gemm_kernel<<<dim3(N_ / 128, R_, 4), 256, 0, stream>>>(supp, tuT, r_sums, z_v);
    gather_kernel<<<(2 * 4096 * H_ + 255) / 256, 256, 0, stream>>>(z_u, z_v, u_idx, v_idx, u_bias, out);
}
